// SentenceAttention_53601191854486
// MI455X (gfx1250) — matrix operand order repaired and checked by reading, not compiled, not run on hardware
//
#include <hip/hip_runtime.h>
#include <math.h>

// Problem constants (match reference: B=256, S=128, E=1024)
#define E_DIM 1024
#define S_DIM 128
#define B_DIM 256
#define M_ROWS (B_DIM * S_DIM)          // 32768 rows of X
#define MT 64                            // rows per GEMM block
#define LSTRIDE 1032                     // LDS row stride in bf16 elems (padding: 8 -> 4-bank shift/row)
#define GEMM_SMEM (2u * MT * LSTRIDE * 2u) // x_hi + x_lo staging, bytes (264192 < 320KB WGP LDS)

typedef __attribute__((ext_vector_type(16))) __bf16       v16bf;
typedef __attribute__((ext_vector_type(8)))  float        v8f;
typedef __attribute__((ext_vector_type(4)))  unsigned int v4u;
typedef __attribute__((ext_vector_type(2)))  unsigned int v2u;

union Frag { v16bf v; v4u q[2]; };

__device__ __forceinline__ unsigned short f32_to_bf16_rne(float f) {
  unsigned int u = __float_as_uint(f);
  unsigned int r = u + 0x7fffu + ((u >> 16) & 1u);   // round-to-nearest-even
  return (unsigned short)(r >> 16);
}
__device__ __forceinline__ float bf16_to_f32(unsigned short h) {
  return __uint_as_float(((unsigned int)h) << 16);
}

// ---------------------------------------------------------------------------
// Kernel 1: split W (fp32) into bf16 hi/lo pair for the 3-term WMMA scheme
// ---------------------------------------------------------------------------
__global__ void prep_w_kernel(const float* __restrict__ W,
                              unsigned short* __restrict__ Whi,
                              unsigned short* __restrict__ Wlo) {
  const int i = blockIdx.x * blockDim.x + threadIdx.x;   // E*E divisible by 256
  const float w = W[i];
  const unsigned short h = f32_to_bf16_rne(w);
  Whi[i] = h;
  Wlo[i] = f32_to_bf16_rne(w - bf16_to_f32(h));
}

// ---------------------------------------------------------------------------
// Kernel 2: fused  scores[m] = sum_f tanh( sum_e x[m,e]*W[f,e] ) * u[f]
// GEMM via v_wmma_f32_16x16x32_bf16 with hi/lo split (~fp32 accuracy).
// Block = 256 threads = 8 waves = 4 M-strips (16 rows) x 2 N-halves.
// X tile (64 rows x 1024) staged in LDS as bf16 hi/lo; W streamed from L2.
// ---------------------------------------------------------------------------
__global__ __launch_bounds__(256) void gemm_tanh_dot_kernel(
    const float* __restrict__ x,
    const unsigned short* __restrict__ Whi,
    const unsigned short* __restrict__ Wlo,
    const float* __restrict__ u,
    float* __restrict__ scores) {
  extern __shared__ unsigned short sm[];
  unsigned short* xhi = sm;
  unsigned short* xlo = sm + MT * LSTRIDE;

  const int t  = threadIdx.x;
  const int m0 = blockIdx.x * MT;

  // ---- Stage 64x1024 fp32 -> bf16 hi/lo into LDS (coalesced float4 loads) ----
  for (int i = 0; i < MT; ++i) {
    const float4 vv = *reinterpret_cast<const float4*>(
        x + (size_t)(m0 + i) * E_DIM + t * 4);
    const unsigned short h0 = f32_to_bf16_rne(vv.x);
    const unsigned short h1 = f32_to_bf16_rne(vv.y);
    const unsigned short h2 = f32_to_bf16_rne(vv.z);
    const unsigned short h3 = f32_to_bf16_rne(vv.w);
    const unsigned short l0 = f32_to_bf16_rne(vv.x - bf16_to_f32(h0));
    const unsigned short l1 = f32_to_bf16_rne(vv.y - bf16_to_f32(h1));
    const unsigned short l2 = f32_to_bf16_rne(vv.z - bf16_to_f32(h2));
    const unsigned short l3 = f32_to_bf16_rne(vv.w - bf16_to_f32(h3));
    const int base = i * LSTRIDE + t * 4;               // 8B aligned
    *reinterpret_cast<v2u*>(xhi + base) =
        (v2u){ (unsigned)h0 | ((unsigned)h1 << 16), (unsigned)h2 | ((unsigned)h3 << 16) };
    *reinterpret_cast<v2u*>(xlo + base) =
        (v2u){ (unsigned)l0 | ((unsigned)l1 << 16), (unsigned)l2 | ((unsigned)l3 << 16) };
  }
  __syncthreads();

  const int wave  = t >> 5;
  const int lane  = t & 31;
  const int strip = wave >> 1;          // 0..3 -> rows [m0+16*strip, +16)
  const int nhalf = wave & 1;           // 0/1  -> N tiles [32*nhalf, +32)

  // A fragment addressing per ISA 16-bit 16x32 layout:
  //   lanes 0-15: row M=lane,   V0..3 = K 0..7,  V4..7 = K 16..23
  //   lanes16-31: row M=lane-16,V0..3 = K 8..15, V4..7 = K 24..31
  const int arow = strip * 16 + (lane & 15);
  const int sel8 = (lane >> 4) * 8;
  const unsigned short* ah_row = xhi + (size_t)arow * LSTRIDE + sel8;
  const unsigned short* al_row = xlo + (size_t)arow * LSTRIDE + sel8;

  float acc[8];
#pragma unroll
  for (int r = 0; r < 8; ++r) acc[r] = 0.0f;

  for (int nt = 0; nt < 32; ++nt) {
    const int n0   = (nhalf * 32 + nt) * 16;
    // B fragment: column N = W row f; lanes 0-15 hold K 0..7,16..23,
    // lanes 16-31 hold K 8..15,24..31 (same order as A).
    const int bcol = n0 + (lane & 15);
    const int bsel = (lane >> 4) * 8;
    const unsigned short* bh_row = Whi + (size_t)bcol * E_DIM + bsel;
    const unsigned short* bl_row = Wlo + (size_t)bcol * E_DIM + bsel;
    const float uval = u[n0 + (lane & 15)];

    v8f c = {0.f, 0.f, 0.f, 0.f, 0.f, 0.f, 0.f, 0.f};
#pragma unroll 4
    for (int k = 0; k < E_DIM; k += 32) {
      Frag ah, al, bh, bl;
      ah.q[0] = *reinterpret_cast<const v4u*>(ah_row + k);
      ah.q[1] = *reinterpret_cast<const v4u*>(ah_row + k + 16);
      al.q[0] = *reinterpret_cast<const v4u*>(al_row + k);
      al.q[1] = *reinterpret_cast<const v4u*>(al_row + k + 16);
      bh.q[0] = *reinterpret_cast<const v4u*>(bh_row + k);
      bh.q[1] = *reinterpret_cast<const v4u*>(bh_row + k + 16);
      bl.q[0] = *reinterpret_cast<const v4u*>(bl_row + k);
      bl.q[1] = *reinterpret_cast<const v4u*>(bl_row + k + 16);
      // 3-term split-bf16: hi*hi + hi*lo + lo*hi, f32 accumulate
      c = __builtin_amdgcn_wmma_f32_16x16x32_bf16(false, ah.v, false, bh.v,
                                                  (short)0, c, false, false);
      c = __builtin_amdgcn_wmma_f32_16x16x32_bf16(false, ah.v, false, bl.v,
                                                  (short)0, c, false, false);
      c = __builtin_amdgcn_wmma_f32_16x16x32_bf16(false, al.v, false, bh.v,
                                                  (short)0, c, false, false);
    }
    // Epilogue: C[r] is element (M = r + 8*(lane>>4), N = lane&15)
#pragma unroll
    for (int r = 0; r < 8; ++r) acc[r] += tanhf(c[r]) * uval;
  }

  // Reduce over the 16 lanes of each half (each half = distinct M rows)
#pragma unroll
  for (int r = 0; r < 8; ++r) {
    float v = acc[r];
    v += __shfl_xor(v, 1, 32);
    v += __shfl_xor(v, 2, 32);
    v += __shfl_xor(v, 4, 32);
    v += __shfl_xor(v, 8, 32);
    acc[r] = v;
  }
  if ((lane & 15) == 0) {
    const int mrow = m0 + strip * 16 + (lane >> 4) * 8;
#pragma unroll
    for (int r = 0; r < 8; ++r) atomicAdd(&scores[mrow + r], acc[r]);
  }
}

// ---------------------------------------------------------------------------
// Kernel 3: per-bag masked softmax + weighted sum over X (+ keep_attention)
// ---------------------------------------------------------------------------
__global__ __launch_bounds__(256) void softmax_weighted_sum_kernel(
    const float* __restrict__ x,
    const float* __restrict__ scores,
    const int* __restrict__ docidx,
    float* __restrict__ out) {
  const int b = blockIdx.x;
  const int t = threadIdx.x;
  __shared__ float red[S_DIM];
  __shared__ float attn[S_DIM];
  __shared__ float stats[2];

  const int start = docidx[2 * b + 0];
  const int end   = docidx[2 * b + 1];

  float val = -3.0e38f;
  bool  in  = false;
  if (t < S_DIM) {
    in  = (t >= start) && (t < end);
    const float s = scores[(size_t)b * S_DIM + t];
    val = in ? s : -3.0e38f;
    red[t] = val;
  }
  __syncthreads();
  for (int st = 64; st >= 1; st >>= 1) {
    if (t < st) red[t] = fmaxf(red[t], red[t + st]);
    __syncthreads();
  }
  if (t == 0) stats[0] = red[0];
  __syncthreads();

  const float mx = stats[0];
  float p = 0.0f;
  if (t < S_DIM) {
    p = in ? expf(val - mx) : 0.0f;
    red[t] = p;
  }
  __syncthreads();
  for (int st = 64; st >= 1; st >>= 1) {
    if (t < st) red[t] = red[t] + red[t + st];
    __syncthreads();
  }
  if (t == 0) stats[1] = red[0];
  __syncthreads();

  const float inv = 1.0f / stats[1];
  if (t < S_DIM) attn[t] = p * inv;
  __syncthreads();

  // out[b, e0..e0+3] = sum_{s in [start,end)} attn[s] * x[b, s, e]
  float4 acc = {0.f, 0.f, 0.f, 0.f};
  const int e0 = t * 4;
  for (int s = start; s < end; ++s) {
    const float a = attn[s];
    const float4 xv = *reinterpret_cast<const float4*>(
        x + ((size_t)b * S_DIM + s) * E_DIM + e0);
    acc.x += a * xv.x; acc.y += a * xv.y; acc.z += a * xv.z; acc.w += a * xv.w;
  }
  *reinterpret_cast<float4*>(out + (size_t)b * E_DIM + e0) = acc;

  // keep_attention = attn of the last bag, padded with zeros (already zero
  // outside the mask since p==0 there)
  if (b == B_DIM - 1 && t < S_DIM) {
    out[(size_t)B_DIM * E_DIM + t] = attn[t];
  }
}

// ---------------------------------------------------------------------------
extern "C" void kernel_launch(void* const* d_in, const int* in_sizes, int n_in,
                              void* d_out, int out_size, void* d_ws, size_t ws_size,
                              hipStream_t stream) {
  (void)in_sizes; (void)n_in; (void)out_size; (void)ws_size;
  const float* x    = (const float*)d_in[0];          // [B,S,E] fp32
  const float* W    = (const float*)d_in[1];          // [E,E]  fp32
  const float* u    = (const float*)d_in[2];          // [1,E]  fp32
  const int*   didx = (const int*)d_in[3];            // [B,2]  int32
  float* out = (float*)d_out;                         // [B*E] ++ [S] fp32

  // Workspace layout: W_hi (2MB) | W_lo (2MB) | scores (128KB)
  unsigned short* Whi = (unsigned short*)d_ws;
  unsigned short* Wlo = Whi + (size_t)E_DIM * E_DIM;
  float* scores = (float*)(Wlo + (size_t)E_DIM * E_DIM);

  hipMemsetAsync(scores, 0, (size_t)M_ROWS * sizeof(float), stream);
  prep_w_kernel<<<(E_DIM * E_DIM) / 256, 256, 0, stream>>>(W, Whi, Wlo);
  gemm_tanh_dot_kernel<<<M_ROWS / MT, 256, GEMM_SMEM, stream>>>(x, Whi, Wlo, u, scores);
  softmax_weighted_sum_kernel<<<B_DIM, 256, 0, stream>>>(x, scores, didx, out);
}
